// FrameGNN_88837103551260
// MI455X (gfx1250) — matrix operand order repaired and checked by reading, not compiled, not run on hardware
//
#include <hip/hip_runtime.h>
#include <hip/hip_bf16.h>

// CDNA5 / gfx1250: wave32, WMMA 16x16x32 f16, TDM async tile loads.
typedef __attribute__((ext_vector_type(16))) _Float16 v16h;
typedef __attribute__((ext_vector_type(8)))  _Float16 v8h;
typedef __attribute__((ext_vector_type(8)))  float    v8f;
typedef __attribute__((ext_vector_type(4)))  float    f32x4;
typedef __attribute__((ext_vector_type(4)))  unsigned u32x4;
typedef __attribute__((ext_vector_type(8)))  int      i32x8;
typedef __attribute__((ext_vector_type(4)))  int      i32x4;

#define HD 128    // H*C
#define NH 4      // heads
#define NGRP 512
#define KDIM 128  // K of every GEMM in this model
#define BSTRIDE 40  // Bsh row stride in halves (80B: multiple of 16B for b128 LDS reads)

#if __has_builtin(__builtin_amdgcn_tensor_load_to_lds) && __has_builtin(__builtin_amdgcn_s_wait_tensorcnt)
#define HAVE_TDM 1
#else
#define HAVE_TDM 0
#endif

// ---------------- helpers ----------------
static __device__ __forceinline__ float wave_sum(float v) {
#pragma unroll
  for (int o = 16; o > 0; o >>= 1) v += __shfl_xor(v, o, 32);
  return v;
}
static __device__ __forceinline__ unsigned enc_f(float f) {
  unsigned u = __float_as_uint(f);
  return (u & 0x80000000u) ? ~u : (u | 0x80000000u);
}
static __device__ __forceinline__ float dec_f(unsigned u) {
  return (u & 0x80000000u) ? __uint_as_float(u & 0x7FFFFFFFu) : __uint_as_float(~u);
}
static inline unsigned host_enc(float f) {
  union { float f; unsigned u; } c; c.f = f;
  return (c.u & 0x80000000u) ? ~c.u : (c.u | 0x80000000u);
}

// ---------------- generic fill ----------------
__global__ void fill_u32_kernel(unsigned* __restrict__ p, unsigned v, size_t n) {
  size_t i = (size_t)blockIdx.x * blockDim.x + threadIdx.x;
  if (i < n) p[i] = v;
}

// ---------------- WMMA GEMM: C = A(N x 128) * B(128 x Nout) + bias ----------------
// 256 threads = 8 waves; block tile 64 rows x 32 cols; wave grid 4x2, 16x16 per wave.
// A tile staged into LDS (f32) via TDM tensor_load_to_lds when available;
// B tile staged transposed as f16 so each lane's fragment is two contiguous 16B runs.
__global__ void __launch_bounds__(256) wmma_gemm_kernel(
    const float* __restrict__ A, const float* __restrict__ B,
    const float* __restrict__ bias, float* __restrict__ C,
    int N, int Nout) {
  __shared__ __attribute__((aligned(32))) float    Ashf[64 * 32];      // [row][k], 8KB
  __shared__ __attribute__((aligned(16))) _Float16 Bsh[32 * BSTRIDE];  // [n][k] transposed

  const int t = threadIdx.x;
  const int wave = t >> 5, lane = t & 31;
  const int lh = lane & 15, hi = lane >> 4;
  const int waveRow = wave >> 1, waveCol = wave & 1;
  const int rowBase = blockIdx.x * 64;
  const int colBase = blockIdx.y * 32;
  v8f acc = {0.f, 0.f, 0.f, 0.f, 0.f, 0.f, 0.f, 0.f};

#pragma unroll
  for (int k0 = 0; k0 < KDIM; k0 += 32) {
    __syncthreads();  // protect LDS tiles from previous iteration's readers

#if HAVE_TDM
    // ---- Tensor Data Mover: DMA the 64x32 f32 A-tile into LDS ----
    if (wave == 0) {
      unsigned long long ga =
          (unsigned long long)(const void*)(A + (size_t)rowBase * KDIM + k0);
      unsigned lds = (unsigned)(unsigned long long)(void*)Ashf;
      // D# group0: count=1 | lds_addr | global_addr[56:0] | type=2
      u32x4 g0 = {1u, lds, (unsigned)ga,
                  (unsigned)((ga >> 32) & 0x01FFFFFFull) | 0x80000000u};
      // Remaining extents relative to tile start (OOB rows of ragged last block -> zero fill)
      unsigned td0 = (unsigned)(KDIM - k0);      // tensor_dim0
      unsigned td1 = (unsigned)(N - rowBase);    // tensor_dim1
      unsigned long long s0 = (unsigned long long)KDIM;  // tensor_dim0_stride
      i32x8 g1 = {
          (int)0x20000u,                                   // data_size=2 (4B)
          (int)((td0 & 0xFFFFu) << 16),                    // tensor_dim0[15:0] @bits63:48
          (int)((td0 >> 16) | ((td1 & 0xFFFFu) << 16)),    // dim0 hi | dim1 lo
          (int)((td1 >> 16) | (32u << 16)),                // dim1 hi | tile_dim0=32
          (int)64u,                                        // tile_dim1=64, tile_dim2=0
          (int)(unsigned)(s0 & 0xFFFFFFFFull),             // dim0_stride[31:0]
          (int)(unsigned)((s0 >> 32) & 0xFFFFull),         // dim0_stride[47:32] | dim1_stride lo=0
          0};
      i32x4 gz = {0, 0, 0, 0};
#if defined(__clang_major__) && __clang_major__ >= 23
      i32x8 gz8 = {0, 0, 0, 0, 0, 0, 0, 0};
      __builtin_amdgcn_tensor_load_to_lds(g0, g1, gz, gz, gz8, 0);
#else
      __builtin_amdgcn_tensor_load_to_lds(g0, g1, gz, gz, 0);
#endif
    }
#else
    // ---- manual cooperative A load: 2 x float4 per thread (coalesced b128) ----
#pragma unroll
    for (int rep = 0; rep < 2; ++rep) {
      int rL = (t >> 3) + rep * 32;
      int kq = t & 7;
      int row = rowBase + rL;
      if (row >= N) row = N - 1;
      f32x4 v = *(const f32x4*)(A + (size_t)row * KDIM + k0 + kq * 4);
      *(f32x4*)&Ashf[rL * 32 + kq * 4] = v;
    }
#endif

    // ---- cooperative B load (coalesced b128), convert f16, store transposed ----
    {
      int kL = t >> 3, nq = t & 7;
      if (k0 + 32 < KDIM)
        __builtin_prefetch(B + (size_t)(k0 + 32 + kL) * Nout + colBase + nq * 4, 0, 1);
      f32x4 v = *(const f32x4*)(B + (size_t)(k0 + kL) * Nout + colBase + nq * 4);
#pragma unroll
      for (int j = 0; j < 4; ++j)
        Bsh[(nq * 4 + j) * BSTRIDE + kL] = (_Float16)v[j];
    }

#if HAVE_TDM
    if (wave == 0) __builtin_amdgcn_s_wait_tensorcnt(0);
#endif
    __syncthreads();

    // ---- build fragments from LDS (contiguous, 16B/32B aligned) ----
    // A 16x32 f16 layout: lanes0-15 K={0..7,16..23}, lanes16-31 K={8..15,24..31}
    const float* ap = &Ashf[(waveRow * 16 + lh) * 32];
    v8f alo = *(const v8f*)(ap + hi * 8);
    v8f ahi = *(const v8f*)(ap + 16 + hi * 8);
    v16h a;
#pragma unroll
    for (int e = 0; e < 8; ++e) {
      a[e]     = (_Float16)alo[e];
      a[8 + e] = (_Float16)ahi[e];
    }
    // B 32x16 f16 layout: lanes0-15 K={0..7,16..23} (N=lane), lanes16-31 K={8..15,24..31}
    const _Float16* bp = &Bsh[(waveCol * 16 + lh) * BSTRIDE];
    v8h b0 = *(const v8h*)(bp + hi * 8);
    v8h b1 = *(const v8h*)(bp + 16 + hi * 8);
    v16h b = __builtin_shufflevector(b0, b1, 0, 1, 2, 3, 4, 5, 6, 7, 8, 9, 10, 11, 12,
                                     13, 14, 15);
    acc = __builtin_amdgcn_wmma_f32_16x16x32_f16(false, a, false, b, (short)0, acc,
                                                 false, false);
  }

  // ---- store C (f32 16x16 layout: vgpr r -> lanes0-15 M=r, lanes16-31 M=r+8) ----
  int n = colBase + waveCol * 16 + lh;
  float bv = bias ? bias[n] : 0.f;
#pragma unroll
  for (int r8 = 0; r8 < 8; ++r8) {
    int m = rowBase + waveRow * 16 + r8 + hi * 8;
    if (m < N) C[(size_t)m * Nout + n] = acc[r8] + bv;
  }
}

// ---------------- BatchNorm over axis 0 ----------------
__global__ void bn_stats_kernel(const float* __restrict__ x, float* __restrict__ stats, int N) {
  int f = threadIdx.x & (HD - 1);
  int r0 = blockIdx.x * 2 + (threadIdx.x >> 7);
  float s = 0.f, q = 0.f;
  for (int r = r0; r < N; r += gridDim.x * 2) {
    float v = x[(size_t)r * HD + f];
    s += v; q += v * v;
  }
  atomicAdd(&stats[f], s);
  atomicAdd(&stats[HD + f], q);
}
__global__ void bn_finalize_kernel(const float* __restrict__ stats, const float* __restrict__ g,
                                   const float* __restrict__ b, float* __restrict__ prm,
                                   float invN) {
  int f = threadIdx.x;
  float m = stats[f] * invN;
  float var = stats[HD + f] * invN - m * m;
  float sc = g[f] * rsqrtf(var + 1e-5f);
  prm[f] = sc;
  prm[HD + f] = b[f] - m * sc;
}
__global__ void affine_kernel(const float* __restrict__ x, const float* __restrict__ prm,
                              float* __restrict__ y, size_t total, int elu) {
  size_t i = (size_t)blockIdx.x * blockDim.x + threadIdx.x;
  if (i >= total) return;
  int f = (int)(i & (HD - 1));
  float v = x[i] * prm[f] + prm[HD + f];
  if (elu) v = v > 0.f ? v : __expf(v) - 1.f;
  y[i] = v;
}

// ---------------- per-node, per-head attention logits (1 head == 1 wave32) ----------------
__global__ void logits_kernel(const float* __restrict__ hW, const float* __restrict__ as_,
                              const float* __restrict__ ad_, float* __restrict__ ls,
                              float* __restrict__ ld, int N) {
  int n = blockIdx.x, t = threadIdx.x;  // 128 threads = 4 waves = 4 heads
  float hv = hW[(size_t)n * HD + t];
  float s = wave_sum(hv * as_[t]);
  float d = wave_sum(hv * ad_[t]);
  if ((t & 31) == 0) {
    int h = t >> 5;
    ls[n * NH + h] = s;
    ld[n * NH + h] = d;
  }
}

// ---------------- edge softmax / aggregation ----------------
__global__ void edge_max_kernel(const int* __restrict__ src, const int* __restrict__ dst,
                                const float* __restrict__ ls, const float* __restrict__ ld,
                                unsigned* __restrict__ emax, int E, int T) {
  size_t i = (size_t)blockIdx.x * blockDim.x + threadIdx.x;
  if (i >= (size_t)T * NH) return;
  int e = (int)(i >> 2), h = (int)(i & 3);
  int s = e < E ? src[e] : e - E;
  int d = e < E ? dst[e] : e - E;
  float l = ls[s * NH + h] + ld[d * NH + h];
  l = l > 0.f ? l : 0.2f * l;  // leaky_relu(0.2)
  atomicMax(&emax[d * NH + h], enc_f(l));
}
__global__ void edge_exp_kernel(const int* __restrict__ src, const int* __restrict__ dst,
                                const float* __restrict__ ls, const float* __restrict__ ld,
                                const unsigned* __restrict__ emax, float* __restrict__ ex,
                                float* __restrict__ eden, int E, int T) {
  size_t i = (size_t)blockIdx.x * blockDim.x + threadIdx.x;
  if (i >= (size_t)T * NH) return;
  int e = (int)(i >> 2), h = (int)(i & 3);
  int s = e < E ? src[e] : e - E;
  int d = e < E ? dst[e] : e - E;
  float l = ls[s * NH + h] + ld[d * NH + h];
  l = l > 0.f ? l : 0.2f * l;
  float v = __expf(l - dec_f(emax[d * NH + h]));
  ex[i] = v;
  atomicAdd(&eden[d * NH + h], v);
}
__global__ void edge_agg_kernel(const int* __restrict__ src, const int* __restrict__ dst,
                                const float* __restrict__ hW, const float* __restrict__ ex,
                                const float* __restrict__ eden, float* __restrict__ agg,
                                int E, int T) {
  size_t i = (size_t)blockIdx.x * blockDim.x + threadIdx.x;
  if (i >= (size_t)T * HD) return;
  int e = (int)(i >> 7), f = (int)(i & (HD - 1));
  int h = f >> 5;
  int s = e < E ? src[e] : e - E;
  int d = e < E ? dst[e] : e - E;
  float alpha = ex[(size_t)e * NH + h] / (eden[d * NH + h] + 1e-16f);
  atomicAdd(&agg[(size_t)d * HD + f], hW[(size_t)s * HD + f] * alpha);
}

// ---------------- LayerNorm(skip) + residual add (one row per block) ----------------
__global__ void ln_residual_kernel(const float* __restrict__ skp, const float* __restrict__ g,
                                   const float* __restrict__ b, float* __restrict__ h, int N) {
  __shared__ float red[8];
  int n = blockIdx.x, t = threadIdx.x;  // 128 threads
  float v = skp[(size_t)n * HD + t];
  float s = wave_sum(v);
  float q = wave_sum(v * v);
  int w = t >> 5;
  if ((t & 31) == 0) { red[w] = s; red[4 + w] = q; }
  __syncthreads();
  float S = red[0] + red[1] + red[2] + red[3];
  float Q = red[4] + red[5] + red[6] + red[7];
  float m = S * (1.f / HD);
  float var = Q * (1.f / HD) - m * m;
  float r = rsqrtf(var + 1e-5f);
  h[(size_t)n * HD + t] += (v - m) * r * g[t] + b[t];
}

// ---------------- gate head: relu(ln(y64)) @ G2 + g2b (64 threads/row) ----------------
__global__ void gate_kernel(const float* __restrict__ y, const float* __restrict__ g,
                            const float* __restrict__ b, const float* __restrict__ G2,
                            const float* __restrict__ g2b, float* __restrict__ gate, int N) {
  __shared__ float red[4];
  int n = blockIdx.x, t = threadIdx.x;  // 64 threads = 2 waves
  float v = y[(size_t)n * 64 + t];
  float s = wave_sum(v);
  float q = wave_sum(v * v);
  if ((t & 31) == 0) { red[t >> 5] = s; red[2 + (t >> 5)] = q; }
  __syncthreads();
  float m = (red[0] + red[1]) * (1.f / 64.f);
  float var = (red[2] + red[3]) * (1.f / 64.f) - m * m;
  float r = rsqrtf(var + 1e-5f);
  float u = (v - m) * r * g[t] + b[t];
  u = fmaxf(u, 0.f);
  float p = wave_sum(u * G2[t]);
  __syncthreads();
  if ((t & 31) == 0) red[t >> 5] = p;
  __syncthreads();
  if (t == 0) gate[n] = red[0] + red[1] + g2b[0];
}

// ---------------- group (batch) softmax + weighted pooling ----------------
__global__ void group_max_kernel(const float* __restrict__ gate, const int* __restrict__ batch,
                                 unsigned* __restrict__ gmax, int N) {
  int i = blockIdx.x * blockDim.x + threadIdx.x;
  if (i >= N) return;
  atomicMax(&gmax[batch[i]], enc_f(gate[i]));
}
__global__ void group_exp_kernel(const float* __restrict__ gate, const int* __restrict__ batch,
                                 const unsigned* __restrict__ gmax, float* __restrict__ gex,
                                 float* __restrict__ gden, int N) {
  int i = blockIdx.x * blockDim.x + threadIdx.x;
  if (i >= N) return;
  int gidx = batch[i];
  float v = __expf(gate[i] - dec_f(gmax[gidx]));
  gex[i] = v;
  atomicAdd(&gden[gidx], v);
}
__global__ void group_out_kernel(const float* __restrict__ h, const float* __restrict__ gex,
                                 const float* __restrict__ gden, const int* __restrict__ batch,
                                 float* __restrict__ out, int N) {
  size_t i = (size_t)blockIdx.x * blockDim.x + threadIdx.x;
  if (i >= (size_t)N * HD) return;
  int n = (int)(i >> 7), f = (int)(i & (HD - 1));
  int gidx = batch[n];
  float w = gex[n] / (gden[gidx] + 1e-16f);
  atomicAdd(&out[(size_t)gidx * HD + f], h[i] * w);
}

// ---------------- host launcher ----------------
extern "C" void kernel_launch(void* const* d_in, const int* in_sizes, int n_in,
                              void* d_out, int out_size, void* d_ws, size_t ws_size,
                              hipStream_t stream) {
  (void)n_in; (void)out_size; (void)ws_size;
  const float* x       = (const float*)d_in[0];
  const float* inorm_g = (const float*)d_in[1];
  const float* inorm_b = (const float*)d_in[2];
  const float* W1      = (const float*)d_in[3];
  const float* as1     = (const float*)d_in[4];
  const float* ad1     = (const float*)d_in[5];
  const float* bn1_g   = (const float*)d_in[7];
  const float* bn1_b   = (const float*)d_in[8];
  const float* W2      = (const float*)d_in[9];
  const float* as2     = (const float*)d_in[10];
  const float* ad2     = (const float*)d_in[11];
  const float* bn2_g   = (const float*)d_in[13];
  const float* bn2_b   = (const float*)d_in[14];
  const float* Wsm     = (const float*)d_in[15];
  const float* bs      = (const float*)d_in[16];
  const float* ln_g    = (const float*)d_in[17];
  const float* ln_b    = (const float*)d_in[18];
  const float* G1      = (const float*)d_in[19];
  const float* g1b     = (const float*)d_in[20];
  const float* gln_g   = (const float*)d_in[21];
  const float* gln_b   = (const float*)d_in[22];
  const float* G2      = (const float*)d_in[23];
  const float* g2b     = (const float*)d_in[24];
  const int*   eidx    = (const int*)d_in[25];
  const int*   batch   = (const int*)d_in[26];

  const int N = in_sizes[0] / HD;
  const int E = in_sizes[25] / 2;
  const int T = E + N;  // edges + self-loops
  const int* srcp = eidx;
  const int* dstp = eidx + E;

  // workspace carve-out (floats)
  float* base = (float*)d_ws;
  size_t off = 0;
  auto alloc = [&](size_t n) { float* p = base + off; off += n; return p; };
  const size_t nHD = (size_t)N * HD;
  float*    xn   = alloc(nHD);  // normalized x -> later h2 -> hfinal
  float*    hW   = alloc(nHD);  // GEMM out per layer; later gate N x 64
  float*    agg  = alloc(nHD);  // attention aggregation accumulator
  float*    h1   = alloc(nHD);  // layer-1 output
  float*    skp  = alloc(nHD);  // skip GEMM out
  float*    ls   = alloc((size_t)N * NH);
  float*    ld   = alloc((size_t)N * NH);
  float*    exb  = alloc((size_t)T * NH);
  float*    eden = alloc((size_t)N * NH);
  unsigned* emax = (unsigned*)alloc((size_t)N * NH);
  float*    stats = alloc(2 * HD);
  float*    prm   = alloc(2 * HD);
  float*    gate  = alloc((size_t)N);
  float*    gex   = alloc((size_t)N);
  float*    gden  = alloc(NGRP);
  unsigned* gmax  = (unsigned*)alloc(NGRP);

  const int TB = 256;
  auto blocks = [](size_t total, int bsz) { return (unsigned)((total + bsz - 1) / bsz); };
  auto fill = [&](void* p, unsigned v, size_t n) {
    fill_u32_kernel<<<blocks(n, TB), TB, 0, stream>>>((unsigned*)p, v, n);
  };
  const unsigned NEG_INF_ENC = host_enc(-3e38f);
  const float invN = 1.f / (float)N;

  // ---- input BatchNorm ----
  fill(stats, 0u, 2 * HD);
  bn_stats_kernel<<<256, 256, 0, stream>>>(x, stats, N);
  bn_finalize_kernel<<<1, HD, 0, stream>>>(stats, inorm_g, inorm_b, prm, invN);
  affine_kernel<<<blocks(nHD, TB), TB, 0, stream>>>(x, prm, xn, nHD, 0);

  // ---- GAT layer (x2) ----
  auto gat_layer = [&](const float* inb, const float* Wm, const float* as_, const float* ad_,
                       const float* bng, const float* bnb, float* outb) {
    dim3 grid(blocks((size_t)N, 64), HD / 32);
    wmma_gemm_kernel<<<grid, 256, 0, stream>>>(inb, Wm, nullptr, hW, N, HD);
    logits_kernel<<<N, HD, 0, stream>>>(hW, as_, ad_, ls, ld, N);
    fill(emax, NEG_INF_ENC, (size_t)N * NH);
    fill(eden, 0u, (size_t)N * NH);
    fill(agg, 0u, nHD);
    size_t t4 = (size_t)T * NH, tF = (size_t)T * HD;
    edge_max_kernel<<<blocks(t4, TB), TB, 0, stream>>>(srcp, dstp, ls, ld, emax, E, T);
    edge_exp_kernel<<<blocks(t4, TB), TB, 0, stream>>>(srcp, dstp, ls, ld, emax, exb, eden, E, T);
    edge_agg_kernel<<<blocks(tF, TB), TB, 0, stream>>>(srcp, dstp, hW, exb, eden, agg, E, T);
    // per-feature GAT bias cancels exactly inside BatchNorm, so it is omitted.
    fill(stats, 0u, 2 * HD);
    bn_stats_kernel<<<256, 256, 0, stream>>>(agg, stats, N);
    bn_finalize_kernel<<<1, HD, 0, stream>>>(stats, bng, bnb, prm, invN);
    affine_kernel<<<blocks(nHD, TB), TB, 0, stream>>>(agg, prm, outb, nHD, 1);
  };
  gat_layer(xn, W1, as1, ad1, bn1_g, bn1_b, h1);
  gat_layer(h1, W2, as2, ad2, bn2_g, bn2_b, xn);  // h2 lands in xn

  // ---- skip path: h += ln(x @ Ws + bs) ----
  {
    dim3 grid(blocks((size_t)N, 64), HD / 32);
    wmma_gemm_kernel<<<grid, 256, 0, stream>>>(x, Wsm, bs, skp, N, HD);
  }
  ln_residual_kernel<<<N, HD, 0, stream>>>(skp, ln_g, ln_b, xn, N);  // xn = hfinal

  // ---- gate: relu(ln(h @ G1 + g1b)) @ G2 + g2b ----
  {
    dim3 grid(blocks((size_t)N, 64), 64 / 32);
    wmma_gemm_kernel<<<grid, 256, 0, stream>>>(xn, G1, g1b, hW, N, 64);
  }
  gate_kernel<<<N, 64, 0, stream>>>(hW, gln_g, gln_b, G2, g2b, gate, N);

  // ---- group softmax pooling -> (512 x 128) output ----
  fill(gmax, NEG_INF_ENC, NGRP);
  fill(gden, 0u, NGRP);
  group_max_kernel<<<blocks((size_t)N, TB), TB, 0, stream>>>(gate, batch, gmax, N);
  group_exp_kernel<<<blocks((size_t)N, TB), TB, 0, stream>>>(gate, batch, gmax, gex, gden, N);
  fill(d_out, 0u, (size_t)NGRP * HD);
  group_out_kernel<<<blocks(nHD, TB), TB, 0, stream>>>(xn, gex, gden, batch, (float*)d_out, N);
}
